// SpanEncoder_73031623901404
// MI455X (gfx1250) — matrix operand order repaired and checked by reading, not compiled, not run on hardware
//
#include <hip/hip_runtime.h>
#include <hip/hip_bf16.h>

#define H        768
#define NSW      4096
#define NC       16384
#define MAXW     30
#define META     20
#define OUTW     (H + H + META + H)   // 2324 floats per candidate row

typedef __attribute__((ext_vector_type(16))) _Float16 v16h;
typedef __attribute__((ext_vector_type(8)))  float    v8f;

// ---------------------------------------------------------------------------
// Kernel 1: token_attn[t] = dot(encoded[t,:], w) + b  via v_wmma_f32_16x16x32_f16
// One wave per 16-token tile, K=768 swept in 24 steps of 32.
// A (16x32 f16): lane<16 -> M=lane, halves 0..7 = K(kt+0..7), halves 8..15 = K(kt+16..23)
//                lane>=16 -> M=lane-16, K blocks +8 (per ISA 7.12.2 16-bit A layout).
// B (32x16 f16): w replicated across all N columns; lane half adds K offset 8, same k order as A.
// D column 0 (lanes 0 and 16) carries the 16 scores.
// ---------------------------------------------------------------------------
__global__ __launch_bounds__(256) void token_attn_wmma(
    const float* __restrict__ E, const float* __restrict__ w,
    const float* __restrict__ b, float* __restrict__ ta)
{
  const int lane = threadIdx.x & 31;
  const int wave = (blockIdx.x * blockDim.x + threadIdx.x) >> 5;   // tile id, 0..255
  const int m    = lane & 15;
  const int klo  = (lane >> 4) << 3;    // A low-half K offset: 0 or 8
  const int kb   = (lane >> 4) << 3;    // B low-half K offset: 0 or 8
  const float* __restrict__ Erow = E + (size_t)(wave * 16 + m) * H;

  v8f c = {};
  for (int kt = 0; kt < H; kt += 32) {
    const float4 a0 = *(const float4*)(Erow + kt + klo);
    const float4 a1 = *(const float4*)(Erow + kt + klo + 4);
    const float4 a2 = *(const float4*)(Erow + kt + klo + 16);
    const float4 a3 = *(const float4*)(Erow + kt + klo + 20);
    const float4 b0 = *(const float4*)(w + kt + kb);
    const float4 b1 = *(const float4*)(w + kt + kb + 4);
    const float4 b2 = *(const float4*)(w + kt + kb + 16);
    const float4 b3 = *(const float4*)(w + kt + kb + 20);

    v16h A, B;
    A[0]=(_Float16)a0.x; A[1]=(_Float16)a0.y; A[2]=(_Float16)a0.z; A[3]=(_Float16)a0.w;
    A[4]=(_Float16)a1.x; A[5]=(_Float16)a1.y; A[6]=(_Float16)a1.z; A[7]=(_Float16)a1.w;
    A[8]=(_Float16)a2.x; A[9]=(_Float16)a2.y; A[10]=(_Float16)a2.z; A[11]=(_Float16)a2.w;
    A[12]=(_Float16)a3.x; A[13]=(_Float16)a3.y; A[14]=(_Float16)a3.z; A[15]=(_Float16)a3.w;
    B[0]=(_Float16)b0.x; B[1]=(_Float16)b0.y; B[2]=(_Float16)b0.z; B[3]=(_Float16)b0.w;
    B[4]=(_Float16)b1.x; B[5]=(_Float16)b1.y; B[6]=(_Float16)b1.z; B[7]=(_Float16)b1.w;
    B[8]=(_Float16)b2.x; B[9]=(_Float16)b2.y; B[10]=(_Float16)b2.z; B[11]=(_Float16)b2.w;
    B[12]=(_Float16)b3.x; B[13]=(_Float16)b3.y; B[14]=(_Float16)b3.z; B[15]=(_Float16)b3.w;

    c = __builtin_amdgcn_wmma_f32_16x16x32_f16(
        /*neg_a=*/false, A, /*neg_b=*/false, B,
        /*c_mod=*/(short)0, c, /*reuse_a=*/false, /*reuse_b=*/false);
  }

  const float bias = b[0];
  if (lane == 0) {
    #pragma unroll
    for (int i = 0; i < 8; ++i) ta[wave * 16 + i] = c[i] + bias;
  } else if (lane == 16) {
    #pragma unroll
    for (int i = 0; i < 8; ++i) ta[wave * 16 + 8 + i] = c[i] + bias;
  }
}

// ---------------------------------------------------------------------------
// Kernel 2: one wave32 per candidate span.
//  - width table staged into LDS via global_load_async_to_lds_b128 (ASYNCcnt)
//  - masked softmax over <=30 scores with wave32 shuffle reductions
//  - attended = sum_w p[w] * encoded[start+w,:], float4-vectorized
//  - start/end embedding copies fused into the w-loop (rows already loaded)
// ---------------------------------------------------------------------------
__global__ __launch_bounds__(256) void span_kernel(
    const float* __restrict__ E, const int* __restrict__ starts,
    const int* __restrict__ ends, const float* __restrict__ table,
    const float* __restrict__ ta, float* __restrict__ out)
{
  __shared__ float s_table[MAXW * META];   // 600 floats = 150 float4

  const int tid = threadIdx.x;
  if (tid < (MAXW * META) / 4) {
    unsigned      lds_addr = (unsigned)(size_t)(&s_table[0]) + (unsigned)tid * 16u;
    const float*  gaddr    = table + tid * 4;
    asm volatile("global_load_async_to_lds_b128 %0, %1, off"
                 :: "v"(lds_addr), "v"(gaddr) : "memory");
  }
  asm volatile("s_wait_asynccnt 0" ::: "memory");
  __syncthreads();

  const int lane  = tid & 31;
  const int cand  = blockIdx.x * (blockDim.x >> 5) + (tid >> 5);
  const int start = starts[cand];
  const int end   = ends[cand];
  const int width = end - start + 1;            // 1..30

  // masked softmax over token_attn[start..end]
  float sc = (lane < width) ? ta[start + lane] : -3.0e38f;
  float mx = sc;
  #pragma unroll
  for (int off = 16; off > 0; off >>= 1) mx = fmaxf(mx, __shfl_xor(mx, off, 32));
  float e = (lane < width) ? __expf(sc - mx) : 0.0f;
  float s = e;
  #pragma unroll
  for (int off = 16; off > 0; off >>= 1) s += __shfl_xor(s, off, 32);
  const float p = e / s;

  float* __restrict__ orow   = out + (size_t)cand * OUTW;
  float4* __restrict__ o_start = (float4*)orow;            // 2324*4 % 16 == 0
  float4* __restrict__ o_end   = (float4*)(orow + H);
  float4* __restrict__ o_att   = (float4*)(orow + H + H + META);

  float4 acc[6] = {};
  for (int wpos = 0; wpos < width; ++wpos) {
    const float  pw = __shfl(p, wpos, 32);
    const float4* __restrict__ row = (const float4*)(E + (size_t)(start + wpos) * H);
    float4 r[6];
    #pragma unroll
    for (int j = 0; j < 6; ++j) r[j] = row[j * 32 + lane];
    #pragma unroll
    for (int j = 0; j < 6; ++j) {
      acc[j].x += pw * r[j].x;  acc[j].y += pw * r[j].y;
      acc[j].z += pw * r[j].z;  acc[j].w += pw * r[j].w;
    }
    if (wpos == 0) {
      #pragma unroll
      for (int j = 0; j < 6; ++j) o_start[j * 32 + lane] = r[j];
    }
    if (wpos == width - 1) {
      #pragma unroll
      for (int j = 0; j < 6; ++j) o_end[j * 32 + lane] = r[j];
    }
  }

  if (lane < META) orow[H + H + lane] = s_table[(width - 1) * META + lane];
  #pragma unroll
  for (int j = 0; j < 6; ++j) o_att[j * 32 + lane] = acc[j];
}

// ---------------------------------------------------------------------------
extern "C" void kernel_launch(void* const* d_in, const int* in_sizes, int n_in,
                              void* d_out, int out_size, void* d_ws, size_t ws_size,
                              hipStream_t stream) {
  const float* E      = (const float*)d_in[0];
  const int*   starts = (const int*)d_in[1];
  const int*   ends   = (const int*)d_in[2];
  const float* table  = (const float*)d_in[3];
  const float* w      = (const float*)d_in[4];
  const float* b      = (const float*)d_in[5];
  float*       out    = (float*)d_out;
  float*       ta     = (float*)d_ws;   // 4096 floats of scratch

  // 256 tiles of 16 tokens, 8 waves per block -> 32 blocks
  token_attn_wmma<<<NSW / 16 / 8, 256, 0, stream>>>(E, w, b, ta);
  // one wave per candidate, 8 waves per block -> 2048 blocks
  span_kernel<<<NC / 8, 256, 0, stream>>>(E, starts, ends, table, ta, out);
}
